// ScaledDotProductAttention_37469294690352
// MI455X (gfx1250) — matrix operand order repaired and checked by reading, not compiled, not run on hardware
//
#include <hip/hip_runtime.h>
#include <hip/hip_bf16.h>
#include <stdint.h>

typedef __attribute__((ext_vector_type(16))) __bf16 v16bf;
typedef __attribute__((ext_vector_type(8)))  float  v8f;

#define TM    64    // q rows per block
#define NWAVE 4
#define LKCH  128   // Lk chunk staged in LDS
#define QSTR  72    // qts[col][k] row stride (ushorts), k=0..63, padded
#define VSTR  136   // vs[d][l]   row stride (ushorts), l=0..127, padded
#define PSTR  72    // P[row][k]  row stride (ushorts), k=0..63, padded
#define ASTR  34    // ps[row][k] row stride (ushorts), k=0..31, padded

struct Phase0 {
  float ia [TM*32];
  float iab[TM*16];
  float ibb[TM*32];
  float wA [2048];
  float wB [512];
  float wBt[512];
  float wAt[2048];
  unsigned short P[TM*PSTR];     // bf16 projected queries, persists until frag gather
};
struct Passes {
  unsigned short qts[LKCH*QSTR]; // qt chunk, transposed: [lk_local][dk] bf16
  unsigned short vs [64*VSTR];   // v  chunk, transposed: [dv][lk_local] bf16
  unsigned short ps [NWAVE*16*ASTR]; // per-wave attn tile (A-matrix staging)
};
union Smem { Phase0 p0; Passes pp; };

union FragB { v16bf v; unsigned int u[8]; };
union FragC { v8f v; float f[8]; };

__device__ __forceinline__ unsigned short f2bf(float f) {
  unsigned int u = __builtin_bit_cast(unsigned int, f);
  u += 0x7fffu + ((u >> 16) & 1u);        // round-to-nearest-even
  return (unsigned short)(u >> 16);
}

__global__ __launch_bounds__(128)
void lowrank_attn_kernel(const float* __restrict__ qg,
                         const float* __restrict__ wAg,
                         const float* __restrict__ wBg,
                         const float* __restrict__ wAtg,
                         const float* __restrict__ wBtg,
                         const float* __restrict__ qtg,
                         const float* __restrict__ vg,
                         const unsigned char* __restrict__ maskg,
                         float* __restrict__ outO,
                         float* __restrict__ outA)
{
  __shared__ Smem sm;
  const int tid  = threadIdx.x;
  const int lane = tid & 31, wave = tid >> 5;
  const int n    = lane & 15, hi = lane >> 4;
  const int blk  = blockIdx.x;
  const int tq = blk & 15, h = (blk >> 4) & 7, b = blk >> 7;
  const int lq0 = tq * TM;
  const int rowBase = lq0 + wave * 16;

  // ---------------- Phase 0: low-rank projection chain (f32, tiny) ----------
  const float* qsrc = qg + (((size_t)b * 1024 + lq0) * 8 + h) * 64;
  for (int i = tid; i < 2048; i += 128) sm.p0.wA[i]  = wAg [h * 2048 + i];
  for (int i = tid; i < 512;  i += 128) sm.p0.wB[i]  = wBg [h * 512  + i];
  for (int i = tid; i < 512;  i += 128) sm.p0.wBt[i] = wBtg[h * 512  + i];
  for (int i = tid; i < 2048; i += 128) sm.p0.wAt[i] = wAtg[h * 2048 + i];
  __syncthreads();
  for (int i = tid; i < TM * 32; i += 128) {      // IA = q @ W_A   [64x32]
    int r = i >> 5, c = i & 31; float acc = 0.f;
    for (int k = 0; k < 64; ++k) acc += qsrc[r * 512 + k] * sm.p0.wA[k * 32 + c];
    sm.p0.ia[i] = acc;
  }
  __syncthreads();
  for (int i = tid; i < TM * 16; i += 128) {      // IAB = IA @ W_B [64x16]
    int r = i >> 4, c = i & 15; float acc = 0.f;
    for (int j = 0; j < 32; ++j) acc += sm.p0.ia[r * 32 + j] * sm.p0.wB[j * 16 + c];
    sm.p0.iab[i] = acc;
  }
  __syncthreads();
  for (int i = tid; i < TM * 32; i += 128) {      // IBB = IAB @ W_Bt [64x32]
    int r = i >> 5, c = i & 31; float acc = 0.f;
    for (int j = 0; j < 16; ++j) acc += sm.p0.iab[r * 16 + j] * sm.p0.wBt[j * 32 + c];
    sm.p0.ibb[i] = acc;
  }
  __syncthreads();
  for (int i = tid; i < TM * 64; i += 128) {      // P = IBB @ W_At [64x64] -> bf16
    int r = i >> 6, c = i & 63; float acc = 0.f;
    for (int m = 0; m < 32; ++m) acc += sm.p0.ibb[r * 32 + m] * sm.p0.wAt[m * 64 + c];
    sm.p0.P[r * PSTR + c] = f2bf(acc);
  }
  __syncthreads();

  // Gather this wave's persistent A fragments of P (16 rows x K=64)
  FragB aP0, aP1;
  {
    const unsigned int* pU = (const unsigned int*)sm.p0.P;
    const int row = wave * 16 + n;               // A-matrix: M = lane&15
#pragma unroll
    for (int j = 0; j < 8; ++j) {
      int kj = (j < 4 ? 2 * j : 16 + 2 * (j - 4)) + 8 * hi; // doc A layout
      aP0.u[j] = pU[(row * PSTR + kj) >> 1];
      aP1.u[j] = pU[(row * PSTR + kj + 32) >> 1];
    }
  }
  __syncthreads();   // P region dead; union flips to Passes

  const float* qtb = qtg + (((size_t)b * 8 + h) * 64) * 1024;
  const float* vb  = vg  + (((size_t)b * 8 + h) * 1024) * 64;
  const unsigned char* mb = maskg + (size_t)b * 1024 * 1024;

  // ---------------- Pass 1: online row max / sum over all Lk ----------------
  float m_run[8], s_run[8];
#pragma unroll
  for (int r = 0; r < 8; ++r) { m_run[r] = -__builtin_huge_valf(); s_run[r] = 0.f; }

  for (int ch = 0; ch < 8; ++ch) {
    const int lkc = ch * LKCH;
    for (int i = tid; i < 4096; i += 128) {      // stage qt chunk transposed bf16
      int k = i >> 6, cp = (i & 63) * 2;
      float f0 = qtb[(size_t)k * 1024 + lkc + cp];
      float f1 = qtb[(size_t)k * 1024 + lkc + cp + 1];
      sm.pp.qts[(cp    ) * QSTR + k] = f2bf(f0);
      sm.pp.qts[(cp + 1) * QSTR + k] = f2bf(f1);
    }
    __syncthreads();
    const unsigned int* qU = (const unsigned int*)sm.pp.qts;
    for (int t = 0; t < 8; ++t) {
      const int c0 = t * 16;
      FragB b0, b1;
#pragma unroll
      for (int j = 0; j < 8; ++j) {              // doc B layout: same K order as A
        int kk = (j < 4 ? 2 * j : 16 + 2 * (j - 4)) + 8 * hi, base = (c0 + n) * QSTR;
        b0.u[j] = qU[(base + kk) >> 1];
        b1.u[j] = qU[(base + kk + 32) >> 1];
      }
      FragC acc; v8f z = {0.f,0.f,0.f,0.f,0.f,0.f,0.f,0.f};
      acc.v = __builtin_amdgcn_wmma_f32_16x16x32_bf16(false, aP0.v, false, b0.v, (short)0, z,     false, false);
      acc.v = __builtin_amdgcn_wmma_f32_16x16x32_bf16(false, aP1.v, false, b1.v, (short)0, acc.v, false, false);
      const int colg = lkc + c0 + n;
#pragma unroll
      for (int r = 0; r < 8; ++r) {
        int row = rowBase + r + 8 * hi;
        float val = acc.f[r] * 0.125f;           // / TEMPERATURE (sqrt(64))
        if (!mb[(size_t)row * 1024 + colg]) val = -1e9f;
        float mn = fmaxf(m_run[r], val);
        s_run[r] = s_run[r] * __expf(m_run[r] - mn) + __expf(val - mn);
        m_run[r] = mn;
      }
    }
    __syncthreads();
  }
  // cross-lane merge of (max,sum) across the 16 lanes sharing each row
#pragma unroll
  for (int off = 1; off <= 8; off <<= 1) {
#pragma unroll
    for (int r = 0; r < 8; ++r) {
      float mo = __shfl_xor(m_run[r], off, 32);
      float so = __shfl_xor(s_run[r], off, 32);
      float mn = fmaxf(m_run[r], mo);
      s_run[r] = s_run[r] * __expf(m_run[r] - mn) + so * __expf(mo - mn);
      m_run[r] = mn;
    }
  }
  float inv_s[8];
#pragma unroll
  for (int r = 0; r < 8; ++r) inv_s[r] = 1.0f / s_run[r];

  // ---------------- Pass 2: recompute, write attn, O += A @ V ---------------
  FragC oAcc[4];
#pragma unroll
  for (int d = 0; d < 4; ++d)
#pragma unroll
    for (int r = 0; r < 8; ++r) oAcc[d].f[r] = 0.f;

  float* attnB = outA + (((size_t)b * 8 + h) * 1024) * 1024;
  unsigned short* psW = sm.pp.ps + wave * 16 * ASTR;

  for (int ch = 0; ch < 8; ++ch) {
    const int lkc = ch * LKCH;
    for (int i = tid; i < 4096; i += 128) {      // qt chunk (again, L2-hot)
      int k = i >> 6, cp = (i & 63) * 2;
      float f0 = qtb[(size_t)k * 1024 + lkc + cp];
      float f1 = qtb[(size_t)k * 1024 + lkc + cp + 1];
      sm.pp.qts[(cp    ) * QSTR + k] = f2bf(f0);
      sm.pp.qts[(cp + 1) * QSTR + k] = f2bf(f1);
    }
    for (int i = tid; i < 4096; i += 128) {      // v chunk transposed bf16
      int l = i >> 5, dp = (i & 31) * 2;
      float f0 = vb[(size_t)(lkc + l) * 64 + dp];
      float f1 = vb[(size_t)(lkc + l) * 64 + dp + 1];
      sm.pp.vs[(dp    ) * VSTR + l] = f2bf(f0);
      sm.pp.vs[(dp + 1) * VSTR + l] = f2bf(f1);
    }
    __syncthreads();
    const unsigned int* qU = (const unsigned int*)sm.pp.qts;
    const unsigned int* vU = (const unsigned int*)sm.pp.vs;
    for (int t = 0; t < 8; ++t) {
      const int c0 = t * 16;
      FragB b0, b1;
#pragma unroll
      for (int j = 0; j < 8; ++j) {
        int kk = (j < 4 ? 2 * j : 16 + 2 * (j - 4)) + 8 * hi, base = (c0 + n) * QSTR;
        b0.u[j] = qU[(base + kk) >> 1];
        b1.u[j] = qU[(base + kk + 32) >> 1];
      }
      FragC acc; v8f z = {0.f,0.f,0.f,0.f,0.f,0.f,0.f,0.f};
      acc.v = __builtin_amdgcn_wmma_f32_16x16x32_bf16(false, aP0.v, false, b0.v, (short)0, z,     false, false);
      acc.v = __builtin_amdgcn_wmma_f32_16x16x32_bf16(false, aP1.v, false, b1.v, (short)0, acc.v, false, false);
      const int colg = lkc + c0 + n;
#pragma unroll
      for (int r = 0; r < 8; ++r) {
        int row = rowBase + r + 8 * hi;
        float val = acc.f[r] * 0.125f;
        if (!mb[(size_t)row * 1024 + colg]) val = -1e9f;
        float p = __expf(val - m_run[r]) * inv_s[r];
        attnB[(size_t)row * 1024 + colg] = p;            // required output
        psW[(r + 8 * hi) * ASTR + (t & 1) * 16 + n] = f2bf(p); // A staging
      }
      __syncthreads();                                   // uniform; p_s visible
      if (t & 1) {                                       // K=32 of attn ready
        FragB ap;
        const unsigned int* pU = (const unsigned int*)psW;
#pragma unroll
        for (int j = 0; j < 8; ++j) {
          int kj = (j < 4 ? 2 * j : 16 + 2 * (j - 4)) + 8 * hi;
          ap.u[j] = pU[(n * ASTR + kj) >> 1];
        }
        const int lbase = (t - 1) * 16;
#pragma unroll
        for (int dt = 0; dt < 4; ++dt) {
          FragB bv;
#pragma unroll
          for (int j = 0; j < 8; ++j) {
            int kk = (j < 4 ? 2 * j : 16 + 2 * (j - 4)) + 8 * hi;
            bv.u[j] = vU[((dt * 16 + n) * VSTR + lbase + kk) >> 1];
          }
          oAcc[dt].v = __builtin_amdgcn_wmma_f32_16x16x32_bf16(false, ap.v, false, bv.v, (short)0, oAcc[dt].v, false, false);
        }
      }
    }
    __syncthreads();
  }

  // ---------------- Write output [B,H,Lq,DV] --------------------------------
  float* ob = outO + (((size_t)b * 8 + h) * 1024) * 64;
#pragma unroll
  for (int dt = 0; dt < 4; ++dt)
#pragma unroll
    for (int r = 0; r < 8; ++r) {
      int row = rowBase + r + 8 * hi;
      ob[(size_t)row * 64 + dt * 16 + n] = oAcc[dt].f[r];
    }
}

extern "C" void kernel_launch(void* const* d_in, const int* in_sizes, int n_in,
                              void* d_out, int out_size, void* d_ws, size_t ws_size,
                              hipStream_t stream) {
  (void)in_sizes; (void)n_in; (void)out_size; (void)d_ws; (void)ws_size;
  const float* q    = (const float*)d_in[0];
  const float* wA   = (const float*)d_in[1];
  const float* wB   = (const float*)d_in[2];
  const float* wAt  = (const float*)d_in[3];
  const float* wBt  = (const float*)d_in[4];
  const float* qt   = (const float*)d_in[5];
  const float* v    = (const float*)d_in[6];
  const unsigned char* mask = (const unsigned char*)d_in[8]; // bool mask
  float* outO = (float*)d_out;                               // [4,8,1024,64]
  float* outA = outO + (size_t)4 * 8 * 1024 * 64;            // [4,8,1024,1024]
  hipLaunchKernelGGL(lowrank_attn_kernel, dim3(512), dim3(128), 0, stream,
                     q, wA, wB, wAt, wBt, qt, v, mask, outO, outA);
}
